// lstm_source__25082609008752
// MI455X (gfx1250) — compile-verified, matrix operand order checked by reading, not run on hardware
//
#include <hip/hip_runtime.h>
#include <stdint.h>

#define LAY   4
#define BATCH 4096
#define DIM   1024
#define GDIM  (4 * DIM)

typedef __attribute__((ext_vector_type(16))) __bf16 v16bf;
typedef __attribute__((ext_vector_type(8)))  float  v8f;

__device__ __forceinline__ unsigned short f2bf(float f) {
  // round-to-nearest-even f32 -> bf16
  unsigned int u = __float_as_uint(f);
  unsigned int r = (u + 0x7FFFu + ((u >> 16) & 1u)) >> 16;
  return (unsigned short)r;
}

__device__ __forceinline__ float sigmoidf_(float x) {
  return 1.0f / (1.0f + __expf(-x));
}
__device__ __forceinline__ float tanhf_(float x) {
  float t = __expf(-2.0f * x);
  return (1.0f - t) / (1.0f + t);
}

// CDNA5 async global->LDS copy, 16B per lane, GVS addressing (saddr + voff).
// Tracked by ASYNCcnt (separate from LOADcnt/DScnt).
__device__ __forceinline__ void async_b128(unsigned lds_addr, unsigned voff,
                                           const unsigned short* sbase) {
  asm volatile("global_load_async_to_lds_b128 %0, %1, %2"
               :: "v"(lds_addr), "v"(voff), "s"(sbase)
               : "memory");
}
__device__ __forceinline__ void wait_async0() {
  asm volatile("s_wait_asynccnt 0" ::: "memory");
}

// f32 -> bf16, 4 elements per thread (all our sizes are multiples of 1024)
__global__ void cvt_f32_bf16(const float* __restrict__ in,
                             unsigned short* __restrict__ out, long n) {
  long i = ((long)blockIdx.x * blockDim.x + threadIdx.x) * 4;
  if (i + 3 < n) {
    float4 f = *(const float4*)(in + i);
    ushort4 o = { f2bf(f.x), f2bf(f.y), f2bf(f.z), f2bf(f.w) };
    *(ushort4*)(out + i) = o;
  }
}

// x = bf16(emb[tokens[b]]) ; one block per batch row, 256 thr * 4 elems = D
__global__ void gather_emb(const float* __restrict__ emb,
                           const int* __restrict__ tok,
                           unsigned short* __restrict__ xbf) {
  int b = blockIdx.x;
  int row = tok[b];
  int t = threadIdx.x;
  float4 f = *(const float4*)(emb + (size_t)row * DIM + t * 4);
  ushort4 o = { f2bf(f.x), f2bf(f.y), f2bf(f.z), f2bf(f.w) };
  *(ushort4*)(xbf + (size_t)b * DIM + t * 4) = o;
}

// Fused LSTM layer, block tile 128 rows x 64 cols-per-gate x 4 gates.
// 8 waves: wave (wmh, wn) owns 64 rows x 16 cols for ALL 4 gates
// -> 16 v8f accumulators, 16 WMMAs per 32-wide K step, 1:1 wmma:ds ratio.
// Double-buffered LDS fed by global_load_async_to_lds_b128.
__global__ __launch_bounds__(256)
void lstm_layer_wmma(const unsigned short* __restrict__ xsrc,
                     const unsigned short* __restrict__ hsrc,
                     const unsigned short* __restrict__ Wx,
                     const unsigned short* __restrict__ Wh,
                     const float* __restrict__ c_in,
                     float* __restrict__ h_out,
                     float* __restrict__ c_out,
                     unsigned short* __restrict__ xnext) {
  __shared__ unsigned short sA[2][128 * 32];      // activation tiles [row][k]
  __shared__ unsigned short sW[2][4 * 64 * 32];   // weight tiles [gate][n][k]

  const int tid  = threadIdx.x;
  const int lane = tid & 31;
  const int wv   = tid >> 5;
  const int wn   = wv & 3;                        // 16-col subtile within 64
  const int wmh  = wv >> 2;                       // 64-row half within 128
  const int m0   = blockIdx.y * 128;
  const int n0   = blockIdx.x * 64;

  // ISA 16-bit A/B fragment layout: lanes 0-15 hold K {0..7,16..23},
  // lanes 16-31 hold K {8..15,24..31}
  const int klo  = (lane & 16) ? 8 : 0;
  const int bcol = wn * 16 + (lane & 15);

  // LDS byte addresses (flat low 32 bits == LDS offset)
  const unsigned sA0 = (unsigned)(uintptr_t)&sA[0][0];
  const unsigned sW0 = (unsigned)(uintptr_t)&sW[0][0];

  const unsigned short* actb[2] = { xsrc, hsrc };
  const unsigned short* wtb[2]  = { Wx, Wh };

  v8f acc[4][4] = {};                             // [gate][m-subtile]

  // ---- async tile stage for K-step t (t in [0,64): phase = t>>5) ----
  auto issue = [&](int t) {
    int phase = t >> 5;
    int k0 = (t & 31) << 5;
    int p = t & 1;
    const unsigned short* act = actb[phase];
    const unsigned short* Wt  = wtb[phase];
    // A tile: 128x32 bf16 = 8KB -> 2 chunks of 16B per thread
#pragma unroll
    for (int i = 0; i < 2; ++i) {
      int ch  = tid + i * 256;                    // 0..511
      int row = ch >> 2;
      int kc  = (ch & 3) << 3;
      unsigned lds  = sA0 + (unsigned)((p * (128 * 32) + row * 32 + kc) * 2);
      unsigned voff = (unsigned)(((m0 + row) * DIM + k0 + kc) * 2);
      async_b128(lds, voff, act);
    }
    // W tile: 4x64x32 bf16 = 16KB -> 4 chunks of 16B per thread
#pragma unroll
    for (int i = 0; i < 4; ++i) {
      int ch  = tid + i * 256;                    // 0..1023
      int g   = ch >> 8;
      int rem = ch & 255;
      int n   = rem >> 2;
      int kc  = (rem & 3) << 3;
      unsigned lds  = sW0 + (unsigned)((p * (4 * 64 * 32) + (g * 64 + n) * 32 + kc) * 2);
      unsigned voff = (unsigned)(((g * DIM + n0 + n) * DIM + k0 + kc) * 2);
      async_b128(lds, voff, Wt);
    }
  };

  issue(0);
  wait_async0();
  __syncthreads();

#pragma unroll 1
  for (int t = 0; t < 64; ++t) {
    if (t < 63) issue(t + 1);                     // prefetch next tile (async)

    const int p = t & 1;
    const unsigned short* pA = &sA[p][0];
    const unsigned short* pW = &sW[p][0];

    union Frag { uint4 q[2]; v16bf v; };
    Frag fa[4], fb;
#pragma unroll
    for (int ms = 0; ms < 4; ++ms) {
      int arow = wmh * 64 + ms * 16 + (lane & 15);
      fa[ms].q[0] = *(const uint4*)(pA + arow * 32 + klo);
      fa[ms].q[1] = *(const uint4*)(pA + arow * 32 + 16 + klo);
    }
#pragma unroll
    for (int g = 0; g < 4; ++g) {
      fb.q[0] = *(const uint4*)(pW + (g * 64 + bcol) * 32 + klo);
      fb.q[1] = *(const uint4*)(pW + (g * 64 + bcol) * 32 + 16 + klo);
#pragma unroll
      for (int ms = 0; ms < 4; ++ms) {
        acc[g][ms] = __builtin_amdgcn_wmma_f32_16x16x32_bf16(
            false, fa[ms].v, false, fb.v, (short)0, acc[g][ms], false, false);
      }
    }

    wait_async0();                                // next tile landed in LDS
    __syncthreads();                              // all waves done with tile t
  }

  // fused LSTM cell epilogue: the 4 gate values for each (b,d) sit in the
  // same lane/VGPR slot across acc[0..3][ms].
#pragma unroll
  for (int ms = 0; ms < 4; ++ms) {
#pragma unroll
    for (int r = 0; r < 8; ++r) {
      int row = wmh * 64 + ms * 16 + r + ((lane & 16) ? 8 : 0);
      int b = m0 + row;
      int d = n0 + bcol;
      size_t off = (size_t)b * DIM + d;
      float ig = sigmoidf_(acc[0][ms][r]);
      float it = tanhf_(acc[1][ms][r]);
      float fg = sigmoidf_(acc[2][ms][r]);
      float og = sigmoidf_(acc[3][ms][r]);
      float cn = fg * c_in[off] + ig * it;
      float hn = og * tanhf_(cn);
      c_out[off] = cn;
      h_out[off] = hn;
      xnext[off] = f2bf(hn);                      // next layer's input, bf16
    }
  }
}

extern "C" void kernel_launch(void* const* d_in, const int* in_sizes, int n_in,
                              void* d_out, int out_size, void* d_ws, size_t ws_size,
                              hipStream_t stream) {
  (void)in_sizes; (void)n_in; (void)out_size; (void)ws_size;
  const float* hs     = (const float*)d_in[0];   // [L,B,D]
  const float* cs     = (const float*)d_in[1];   // [L,B,D]
  const int*   tokens = (const int*)d_in[2];     // [B]
  const float* emb    = (const float*)d_in[3];   // [V,D]
  const float* Wx     = (const float*)d_in[4];   // [L,4D,D]
  const float* Wh     = (const float*)d_in[5];   // [L,4D,D]
  float* out = (float*)d_out;                    // [2,L,B,D]: hs then cs

  const size_t WEL = (size_t)LAY * GDIM * DIM;   // 16,777,216 per weight tensor
  const size_t BD  = (size_t)BATCH * DIM;        // 4,194,304

  unsigned short* ws   = (unsigned short*)d_ws;
  unsigned short* Wxbf = ws;
  unsigned short* Whbf = Wxbf + WEL;
  unsigned short* xb0  = Whbf + WEL;
  unsigned short* xb1  = xb0 + BD;
  unsigned short* hbf  = xb1 + BD;               // total ~92 MB of d_ws

  // one-pass f32 -> bf16 of weights (per-layer bf16 weights then live in L2)
  cvt_f32_bf16<<<(int)(WEL / 1024), 256, 0, stream>>>(Wx, Wxbf, (long)WEL);
  cvt_f32_bf16<<<(int)(WEL / 1024), 256, 0, stream>>>(Wh, Whbf, (long)WEL);
  gather_emb<<<BATCH, 256, 0, stream>>>(emb, tokens, xb0);

  unsigned short* xcur = xb0;
  unsigned short* xnxt = xb1;
  for (int l = 0; l < LAY; ++l) {
    cvt_f32_bf16<<<(int)(BD / 1024), 256, 0, stream>>>(hs + (size_t)l * BD, hbf, (long)BD);
    dim3 grid(DIM / 64, BATCH / 128);            // 16 x 32 blocks
    lstm_layer_wmma<<<grid, 256, 0, stream>>>(
        xcur, hbf,
        Wxbf + (size_t)l * GDIM * DIM,
        Whbf + (size_t)l * GDIM * DIM,
        cs + (size_t)l * BD,
        out + (size_t)l * BD,                    // next_hs[l]
        out + (size_t)(LAY + l) * BD,            // next_cs[l]
        xnxt);
    unsigned short* t = xcur; xcur = xnxt; xnxt = t;
  }
}
